// ResNet_cifar10_23218593202476
// MI455X (gfx1250) — 8-bit matrix operand order repaired and checked by reading, compiled, not run on hardware
//
#include <hip/hip_runtime.h>
#include <hip/hip_bf16.h>
#include <math.h>

typedef __attribute__((ext_vector_type(8))) int v8i;

#define EPS 1e-5f

__global__ void k_conv1(const float* __restrict__ x, const float* __restrict__ w,
                        float* __restrict__ out, int N) {
  int idx = blockIdx.x * blockDim.x + threadIdx.x;
  int total = N * 16 * 32 * 32;
  if (idx >= total) return;
  int wo = idx & 31, ho = (idx >> 5) & 31, co = (idx >> 10) & 15, n = idx >> 14;
  float acc = 0.f;
  for (int ci = 0; ci < 3; ++ci)
    for (int kh = 0; kh < 3; ++kh) {
      int h = ho + kh - 1;
      if (h < 0 || h >= 32) continue;
      for (int kw = 0; kw < 3; ++kw) {
        int wc = wo + kw - 1;
        if (wc < 0 || wc >= 32) continue;
        float wv = w[((co * 3 + ci) * 3 + kh) * 3 + kw];
        float s = (wv > 0.f) ? 1.f : ((wv < 0.f) ? -1.f : 0.f);
        acc += s * x[((n * 3 + ci) * 32 + h) * 32 + wc];
      }
    }
  out[idx] = acc;
}

__global__ void k_bn_stats(const float* __restrict__ x, float* __restrict__ mean,
                           float* __restrict__ var, int N, int C, int HW) {
  int c = blockIdx.x;
  int tid = threadIdx.x;
  int per = N * HW;
  float s = 0.f, s2 = 0.f;
  for (int i = tid; i < per; i += blockDim.x) {
    int n = i / HW, r = i % HW;
    float v = x[((size_t)n * C + c) * HW + r];
    s += v; s2 += v * v;
  }
  __shared__ float sh0[256], sh1[256];
  sh0[tid] = s; sh1[tid] = s2;
  __syncthreads();
  for (int off = 128; off > 0; off >>= 1) {
    if (tid < off) { sh0[tid] += sh0[tid + off]; sh1[tid] += sh1[tid + off]; }
    __syncthreads();
  }
  if (tid == 0) {
    float m = sh0[0] / (float)per;
    mean[c] = m;
    var[c] = sh1[0] / (float)per - m * m;
  }
}

__global__ void k_bn_apply(float* __restrict__ x, const float* __restrict__ mean,
                           const float* __restrict__ var, const float* __restrict__ g,
                           const float* __restrict__ b, int C, int HW, int total) {
  int idx = blockIdx.x * blockDim.x + threadIdx.x;
  if (idx >= total) return;
  int c = (idx / HW) % C;
  x[idx] = (x[idx] - mean[c]) * rsqrtf(var[c] + EPS) * g[c] + b[c];
}

__global__ void k_binact_nhwc(const float* __restrict__ x, signed char* __restrict__ a,
                              int lhw, int lc, int total) {
  int idx = blockIdx.x * blockDim.x + threadIdx.x;
  if (idx >= total) return;
  float v = x[idx];
  signed char s = (v > 0.f) ? (signed char)1 : ((v < 0.f) ? (signed char)-1 : (signed char)0);
  int HWm = (1 << lhw) - 1, Cm = (1 << lc) - 1;
  int r = idx & HWm;
  int t = idx >> lhw;
  int c = t & Cm;
  int n = t >> lc;
  a[((((size_t)n << lhw) + r) << lc) + c] = s;
}

__global__ void k_binact(const float* __restrict__ x, signed char* __restrict__ a, int total) {
  int idx = blockIdx.x * blockDim.x + threadIdx.x;
  if (idx >= total) return;
  float v = x[idx];
  a[idx] = (v > 0.f) ? (signed char)1 : ((v < 0.f) ? (signed char)-1 : (signed char)0);
}

__global__ void k_wprep(const float* __restrict__ w, signed char* __restrict__ w8,
                        int Cout, int Cin, int KK, int Kp, int lc) {
  int idx = blockIdx.x * blockDim.x + threadIdx.x;
  if (idx >= Cout * Kp) return;
  int co = idx / Kp, k = idx - co * Kp;
  int blk = k >> lc, ci = k & (Cin - 1);
  signed char s = 0;
  if (blk < KK) {
    float v = w[((size_t)co * Cin + ci) * KK + blk];
    s = (v > 0.f) ? 1 : ((v < 0.f) ? -1 : 0);
  }
  w8[(size_t)co * Kp + k] = s;
}

#define MAX_KP 2304
template <int KH, int KW, bool PADK>
__global__ void k_bingemm(const signed char* __restrict__ A8,
                          const signed char* __restrict__ B8,
                          float* __restrict__ out,
                          int Cout, int Kp, int lc,
                          int Hin, int Win, int Ho, int Wo,
                          int stride, int pad, int NtGB) {
  __shared__ signed char shA[16 * (MAX_KP + 16)];
  const int ldA = Kp + 16;

  int tM = blockIdx.x / NtGB;
  int wid = threadIdx.x >> 5;
  int tG = (blockIdx.x % NtGB) * 8 + wid;
  int lane = threadIdx.x & 31;
  int half = lane >> 4, l15 = lane & 15;
  int Cin = 1 << lc;
  int HoWo = Ho * Wo;

  {
    const signed char* Atile = A8 + (size_t)tM * 16 * Kp;
    int tid = threadIdx.x;
    for (int i = tid * 16; i < 16 * Kp; i += 256 * 16) {
      int r = i / Kp, c = i - r * Kp;
      *(int4*)(&shA[r * ldA + c]) = *(const int4*)(Atile + i);
    }
  }
  __syncthreads();

  const signed char* Bn[4];
  int hb[4], wb[4], nn[4], hh[4], ww[4];
#pragma unroll
  for (int j = 0; j < 4; ++j) {
    int col = (tG * 4 + j) * 16 + l15;
    int n = col / HoWo;
    int r = col - n * HoWo;
    int ho = r / Wo, wo = r - ho * Wo;
    nn[j] = n; hh[j] = ho; ww[j] = wo;
    hb[j] = ho * stride - pad;
    wb[j] = wo * stride - pad;
    Bn[j] = B8 + (((size_t)n * Hin * Win) << lc);
  }
  const signed char* Arow = &shA[l15 * ldA];

  v8i z = {0, 0, 0, 0, 0, 0, 0, 0};
  v8i acc[4] = {z, z, z, z};

  for (int k0 = 0; k0 < Kp; k0 += 64) {
    v8i a;
#pragma unroll
    for (int v = 0; v < 8; ++v) {
      int kb = ((v >> 2) << 5) + (half << 4) + ((v & 3) << 2);
      a[v] = *(const int*)(Arow + k0 + kb);
    }
#pragma unroll
    for (int j = 0; j < 4; ++j) {
      v8i b;
#pragma unroll
      for (int v = 0; v < 8; ++v) {
        int k = k0 + ((v >> 2) << 5) + (half << 4) + ((v & 3) << 2);
        int blk = k >> lc;
        int ci = k & (Cin - 1);
        int val = 0;
        bool ok = true;
        if (PADK) ok = (blk < KH * KW);
        if (ok) {
          int kh = blk / KW, kw = blk - kh * KW;
          int h = hb[j] + kh, w = wb[j] + kw;
          if (h >= 0 && h < Hin && w >= 0 && w < Win)
            val = *(const int*)(Bn[j] + (((size_t)h * Win + w) << lc) + ci);
        }
        b[v] = val;
      }
      acc[j] = __builtin_amdgcn_wmma_i32_16x16x64_iu8(true, a, true, b, acc[j], false, false);
    }
  }
#pragma unroll
  for (int j = 0; j < 4; ++j)
#pragma unroll
    for (int v = 0; v < 8; ++v) {
      int m = tM * 16 + v + half * 8;
      out[((size_t)(nn[j] * Cout + m) * Ho + hh[j]) * Wo + ww[j]] = (float)acc[j][v];
    }
}

__global__ void k_add(float* __restrict__ x, const float* __restrict__ r, int total) {
  int idx = blockIdx.x * blockDim.x + threadIdx.x;
  if (idx < total) x[idx] += r[idx];
}

__global__ void k_avgpool(const float* __restrict__ x, float* __restrict__ pool, int N) {
  int idx = blockIdx.x * blockDim.x + threadIdx.x;
  if (idx >= N * 256) return;
  const float* p = x + (size_t)idx * 64;
  float s = 0.f;
  for (int i = 0; i < 64; ++i) s += p[i];
  pool[idx] = s * (1.f / 64.f);
}

__global__ void k_fcwprep(const float* __restrict__ w, signed char* __restrict__ w8) {
  int idx = blockIdx.x * blockDim.x + threadIdx.x;
  if (idx >= 16 * 256) return;
  int cls = idx >> 8, k = idx & 255;
  signed char s = 0;
  if (cls < 10) {
    float v = w[cls * 256 + k];
    s = (v > 0.f) ? 1 : ((v < 0.f) ? -1 : 0);
  }
  w8[idx] = s;
}

__global__ void k_fcgemm(const signed char* __restrict__ A8,
                         const signed char* __restrict__ W8,
                         const float* __restrict__ bias, float* __restrict__ out) {
  int tM = blockIdx.x;
  int lane = threadIdx.x & 31;
  int half = lane >> 4, l15 = lane & 15;
  const signed char* Abase = A8 + (size_t)(tM * 16 + l15) * 256;
  const signed char* Wrow = W8 + (size_t)l15 * 256;
  v8i acc = {0, 0, 0, 0, 0, 0, 0, 0};
  for (int k0 = 0; k0 < 256; k0 += 64) {
    v8i a, b;
#pragma unroll
    for (int v = 0; v < 8; ++v) {
      int kb = ((v >> 2) << 5) + (half << 4) + ((v & 3) << 2);
      a[v] = *(const int*)(Abase + k0 + kb);
    }
#pragma unroll
    for (int v = 0; v < 8; ++v) {
      int kb = k0 + ((v >> 2) << 5) + (half << 4) + ((v & 3) << 2);
      b[v] = *(const int*)(Wrow + kb);
    }
    acc = __builtin_amdgcn_wmma_i32_16x16x64_iu8(true, a, true, b, acc, false, false);
  }
#pragma unroll
  for (int v = 0; v < 8; ++v) {
    int row = tM * 16 + v + half * 8;
    float bv = (l15 < 10) ? bias[l15] : 0.f;
    out[row * 16 + l15] = (float)acc[v] + bv;
  }
}

__global__ void k_bn1d_stats(const float* __restrict__ fc, float* __restrict__ mean,
                             float* __restrict__ var) {
  int c = threadIdx.x;
  if (c >= 10) return;
  float s = 0.f, s2 = 0.f;
  for (int b = 0; b < 256; ++b) {
    float v = fc[b * 16 + c];
    s += v; s2 += v * v;
  }
  float m = s / 256.f;
  mean[c] = m;
  var[c] = s2 / 256.f - m * m;
}

__global__ void k_final(const float* __restrict__ fc, const float* __restrict__ mean,
                        const float* __restrict__ var, const float* __restrict__ g,
                        const float* __restrict__ b, float* __restrict__ out) {
  int n = threadIdx.x;
  float z[10];
  float mx = -1e30f;
  for (int c = 0; c < 10; ++c) {
    float v = (fc[n * 16 + c] - mean[c]) * rsqrtf(var[c] + EPS) * g[c] + b[c];
    z[c] = v;
    mx = fmaxf(mx, v);
  }
  float s = 0.f;
  for (int c = 0; c < 10; ++c) s += expf(z[c] - mx);
  float ls = mx + logf(s);
  for (int c = 0; c < 10; ++c) out[n * 10 + c] = z[c] - ls;
}

static int ilog2(int v) { int l = 0; while ((1 << l) < v) ++l; return l; }

static void run_bin_conv(const float* wdev, int Cout, int Cin, int KH, int KW,
                         int stride, int pad, const signed char* actNHWC,
                         int N, int Hin, float* outbuf, signed char* w8,
                         hipStream_t stream) {
  int lc = ilog2(Cin);
  int KK = KH * KW;
  int K = Cin * KK;
  int Kp = (K + 63) & ~63;
  int totw = Cout * Kp;
  k_wprep<<<(totw + 255) / 256, 256, 0, stream>>>(wdev, w8, Cout, Cin, KK, Kp, lc);
  int Ho = (Hin + 2 * pad - KH) / stride + 1;
  int Ntiles = N * Ho * Ho / 16;
  int NtGB = Ntiles / 4 / 8;
  int blocks = (Cout / 16) * NtGB;
  if (KH == 3) {
    if (Kp != K)
      k_bingemm<3, 3, true><<<blocks, 256, 0, stream>>>(w8, actNHWC, outbuf, Cout, Kp, lc,
                                                        Hin, Hin, Ho, Ho, stride, pad, NtGB);
    else
      k_bingemm<3, 3, false><<<blocks, 256, 0, stream>>>(w8, actNHWC, outbuf, Cout, Kp, lc,
                                                         Hin, Hin, Ho, Ho, stride, pad, NtGB);
  } else {
    k_bingemm<1, 1, false><<<blocks, 256, 0, stream>>>(w8, actNHWC, outbuf, Cout, Kp, lc,
                                                       Hin, Hin, Ho, Ho, stride, pad, NtGB);
  }
}

static void run_bn(float* x, int N, int C, int HW, const float* g, const float* b,
                   float* stats, hipStream_t stream) {
  k_bn_stats<<<C, 256, 0, stream>>>(x, stats, stats + 512, N, C, HW);
  int total = N * C * HW;
  k_bn_apply<<<(total + 255) / 256, 256, 0, stream>>>(x, stats, stats + 512, g, b, C, HW, total);
}

extern "C" void kernel_launch(void* const* d_in, const int* in_sizes, int n_in,
                              void* d_out, int out_size, void* d_ws, size_t ws_size,
                              hipStream_t stream) {
  (void)in_sizes; (void)n_in; (void)out_size; (void)ws_size;
  const int N = 256;
  const float* in[71];
  for (int i = 0; i < 71; ++i) in[i] = (const float*)d_in[i];

  char* ws = (char*)d_ws;
  float* buf0        = (float*)(ws);
  float* buf1        = (float*)(ws + 67108864UL);
  float* bufR        = (float*)(ws + 134217728UL);
  signed char* bufA  = (signed char*)(ws + 201326592UL);
  signed char* bufW  = (signed char*)(ws + 218103808UL);
  float* stats       = (float*)(ws + 219152384UL);
  float* pool        = (float*)(ws + 219156480UL);
  signed char* poolA = (signed char*)(ws + 219418624UL);
  signed char* fcw8  = (signed char*)(ws + 219484160UL);
  float* fcout       = (float*)(ws + 219488256UL);

  {
    int total = N * 16 * 32 * 32;
    k_conv1<<<(total + 255) / 256, 256, 0, stream>>>(in[0], in[1], buf0, N);
    run_bn(buf0, N, 16, 32 * 32, in[2], in[3], stats, stream);
  }

  struct Blk { int base; int planes; int stride; int hasDs; };
  const Blk blocks[9] = {
    {4, 64, 1, 1},  {13, 64, 1, 0},  {19, 64, 1, 0},
    {25, 128, 2, 1}, {34, 128, 1, 0}, {40, 128, 1, 0},
    {46, 256, 2, 1}, {55, 256, 1, 0}, {61, 256, 1, 0},
  };

  float* cur = buf0;
  float* other = buf1;
  int curC = 16, curH = 32;

  for (int bi = 0; bi < 9; ++bi) {
    const Blk& B = blocks[bi];
    int Cin = curC, Hin = curH, st = B.stride, P = B.planes;
    int Ho = (Hin + 2 - 3) / st + 1;
    int inTot = N * Cin * Hin * Hin;
    k_binact_nhwc<<<(inTot + 255) / 256, 256, 0, stream>>>(
        cur, bufA, ilog2(Hin * Hin), ilog2(Cin), inTot);
    if (B.hasDs) {
      run_bin_conv(in[B.base + 6], P, Cin, 1, 1, st, 0, bufA, N, Hin, bufR, bufW, stream);
      run_bn(bufR, N, P, Ho * Ho, in[B.base + 7], in[B.base + 8], stats, stream);
    }
    run_bin_conv(in[B.base + 0], P, Cin, 3, 3, st, 1, bufA, N, Hin, other, bufW, stream);
    run_bn(other, N, P, Ho * Ho, in[B.base + 1], in[B.base + 2], stats, stream);
    int midTot = N * P * Ho * Ho;
    k_binact_nhwc<<<(midTot + 255) / 256, 256, 0, stream>>>(
        other, bufA, ilog2(Ho * Ho), ilog2(P), midTot);
    run_bin_conv(in[B.base + 3], P, P, 3, 3, 1, 1, bufA, N, Ho, other, bufW, stream);
    run_bn(other, N, P, Ho * Ho, in[B.base + 4], in[B.base + 5], stats, stream);
    k_add<<<(midTot + 255) / 256, 256, 0, stream>>>(other, B.hasDs ? bufR : cur, midTot);
    float* t = cur; cur = other; other = t;
    curC = P; curH = Ho;
  }

  {
    int tot = N * 256;
    k_avgpool<<<(tot + 255) / 256, 256, 0, stream>>>(cur, pool, N);
    k_binact<<<(tot + 255) / 256, 256, 0, stream>>>(pool, poolA, tot);
    k_fcwprep<<<(16 * 256 + 255) / 256, 256, 0, stream>>>(in[67], fcw8);
    k_fcgemm<<<16, 32, 0, stream>>>(poolA, fcw8, in[68], fcout);
    k_bn1d_stats<<<1, 32, 0, stream>>>(fcout, stats, stats + 512);
    k_final<<<1, 256, 0, stream>>>(fcout, stats, stats + 512, in[69], in[70], (float*)d_out);
  }
}
